// ExaoneFlashAttention_82154134438142
// MI455X (gfx1250) — matrix operand order repaired and checked by reading, not compiled, not run on hardware
//
#include <hip/hip_runtime.h>

// ---------------- problem constants ----------------
#define BB   2
#define SS   2048
#define DD   2048
#define HH   16
#define HDIM 128
#define CAPV 50.0f
#define SCALEV 0.088388347648318440550f   // 1/sqrt(128)

typedef __attribute__((ext_vector_type(16))) __bf16 v16bf;
typedef __attribute__((ext_vector_type(8)))  float  v8f;
typedef int v4i __attribute__((vector_size(16)));   // matches builtin param type

union FragU { v16bf f; uint4 q[2]; };

__device__ __forceinline__ v8f wmma_bf16(v16bf a, v16bf b, v8f c) {
    return __builtin_amdgcn_wmma_f32_16x16x32_bf16(false, a, false, b, (short)0, c, false, false);
}

__device__ __forceinline__ float bf2f(__bf16 x) {
    unsigned short u = __builtin_bit_cast(unsigned short, x);
    unsigned v = ((unsigned)u) << 16;
    return __builtin_bit_cast(float, v);
}
__device__ __forceinline__ __bf16 f2bf(float f) {
    unsigned u = __builtin_bit_cast(unsigned, f);
    unsigned r = u + 0x7FFFu + ((u >> 16) & 1u);   // round-to-nearest-even
    unsigned short s = (unsigned short)(r >> 16);
    return __builtin_bit_cast(__bf16, s);
}

// fast tanh via v_exp_f32: tanh(x) = (e^2x - 1)/(e^2x + 1)
__device__ __forceinline__ float fast_tanh(float x) {
    x = fminf(fmaxf(x, -15.0f), 15.0f);
    float e = __expf(2.0f * x);
    return (e - 1.0f) / (e + 1.0f);
}

// ---------------- CDNA5 async copy (global -> LDS), guarded ----------------
#if defined(__has_builtin)
#if __has_builtin(__builtin_amdgcn_global_load_async_to_lds_b128) && \
    __has_builtin(__builtin_amdgcn_s_wait_asynccnt)
#define HAVE_ASYNC_LDS 1
#endif
#endif

__device__ __forceinline__ void cp_async_b128(void* lds, const void* g) {
#ifdef HAVE_ASYNC_LDS
    __builtin_amdgcn_global_load_async_to_lds_b128(
        (__attribute__((address_space(1))) v4i*)(g),
        (__attribute__((address_space(3))) v4i*)(lds), 0, 0);
#else
    *(uint4*)lds = *(const uint4*)g;
#endif
}
__device__ __forceinline__ void cp_async_wait() {
#ifdef HAVE_ASYNC_LDS
    __builtin_amdgcn_s_wait_asynccnt(0);
#endif
}

// ---------------- conversion kernels ----------------
__global__ __launch_bounds__(256) void cvt_f32_bf16(const float* __restrict__ src,
                                                    __bf16* __restrict__ dst, long n) {
    long i = (long)blockIdx.x * blockDim.x + threadIdx.x;
    if (i < n) dst[i] = f2bf(src[i]);
}

// dst[n][k] = src[k][n], D x D, fp32 -> bf16
__global__ __launch_bounds__(256) void transpose_cvt(const float* __restrict__ src,
                                                     __bf16* __restrict__ dst) {
    long i = (long)blockIdx.x * blockDim.x + threadIdx.x;
    long total = (long)DD * DD;
    if (i >= total) return;
    int nrow = (int)(i / DD);
    int k    = (int)(i % DD);
    dst[i] = f2bf(src[(long)k * DD + nrow]);
}

// ---------------- bf16 WMMA GEMM:  C[M][N] = A[M][K] * Bt[N][K]^T ----------------
// block tile 128x256, 8 waves (256 thr), wave tile 64x64 (4x4 of 16x16)
// double-buffered LDS, async global->LDS staging
template<bool F32OUT>
__global__ __launch_bounds__(256) void gemm_bf16_nt(const __bf16* __restrict__ A,
                                                    const __bf16* __restrict__ Bt,
                                                    void* __restrict__ Cout,
                                                    int M, int N, int K) {
    __shared__ __align__(16) __bf16 As[2][128][40];   // 20.5 KB
    __shared__ __align__(16) __bf16 Bs[2][256][40];   // 41 KB

    const int tid  = threadIdx.x;
    const int wave = tid >> 5, lane = tid & 31;
    const int g = lane >> 4, nn = lane & 15;
    const int wm = wave & 1, wn = wave >> 1;            // wave at (wm*64, wn*64)
    const int bm = blockIdx.y * 128, bn = blockIdx.x * 256;

    v8f acc[4][4] = {};

    // A staging: thread t loads row t>>1, 16 elements at col (t&1)*16 (2 x b128)
    const int ar = tid >> 1, ac = (tid & 1) * 16;
    // B staging: thread t loads row t, 32 elements (4 x b128)
    const int br = tid;

    const __bf16* Abase = A  + (long)(bm + ar) * K + ac;
    const __bf16* Bbase = Bt + (long)(bn + br) * K;

    // prologue: stage k0 = 0 into buffer 0
    {
        cp_async_b128(&As[0][ar][ac],     Abase);
        cp_async_b128(&As[0][ar][ac + 8], Abase + 8);
#pragma unroll
        for (int i = 0; i < 4; ++i)
            cp_async_b128(&Bs[0][br][i * 8], Bbase + i * 8);
    }
    cp_async_wait();
    __syncthreads();

    for (int k0 = 0; k0 < K; k0 += 32) {
        const int cur = (k0 >> 5) & 1;
        const bool more = (k0 + 32) < K;
        if (more) {
            const __bf16* an = Abase + k0 + 32;
            const __bf16* bnp = Bbase + k0 + 32;
            cp_async_b128(&As[cur ^ 1][ar][ac],     an);
            cp_async_b128(&As[cur ^ 1][ar][ac + 8], an + 8);
#pragma unroll
            for (int i = 0; i < 4; ++i)
                cp_async_b128(&Bs[cur ^ 1][br][i * 8], bnp + i * 8);
        }

        FragU af[4], bf_[4];
#pragma unroll
        for (int mi = 0; mi < 4; ++mi) {
            const __bf16* r = &As[cur][wm * 64 + mi * 16 + nn][0];
            af[mi].q[0] = *(const uint4*)(r + 8 * g);
            af[mi].q[1] = *(const uint4*)(r + 16 + 8 * g);
        }
#pragma unroll
        for (int ni = 0; ni < 4; ++ni) {
            const __bf16* r = &Bs[cur][wn * 64 + ni * 16 + nn][0];
            bf_[ni].q[0] = *(const uint4*)(r + 8 * g);
            bf_[ni].q[1] = *(const uint4*)(r + 16 + 8 * g);
        }
#pragma unroll
        for (int mi = 0; mi < 4; ++mi)
#pragma unroll
            for (int ni = 0; ni < 4; ++ni)
                acc[mi][ni] = wmma_bf16(af[mi].f, bf_[ni].f, acc[mi][ni]);

        if (more) cp_async_wait();
        __syncthreads();
    }

#pragma unroll
    for (int mi = 0; mi < 4; ++mi)
#pragma unroll
        for (int ni = 0; ni < 4; ++ni)
#pragma unroll
            for (int v = 0; v < 8; ++v) {
                int r = bm + wm * 64 + mi * 16 + v + 8 * g;
                int c = bn + wn * 64 + ni * 16 + nn;
                if (F32OUT)
                    ((float*)Cout)[(long)r * N + c] = acc[mi][ni][v];
                else
                    ((__bf16*)Cout)[(long)r * N + c] = f2bf(acc[mi][ni][v]);
            }
}

// ---------------- RoPE + layout pack ----------------
// q,k,v: [B*S][D] bf16 (head h occupies cols h*128..h*128+127)
// Qr,Kr: [B*H][S][HD] bf16 (RoPE applied); Vt: [B*H][HD][S] bf16
__global__ __launch_bounds__(256) void rope_pack(const __bf16* __restrict__ q,
                                                 const __bf16* __restrict__ k,
                                                 const __bf16* __restrict__ v,
                                                 __bf16* __restrict__ Qr,
                                                 __bf16* __restrict__ Kr,
                                                 __bf16* __restrict__ Vt) {
    long idx = (long)blockIdx.x * blockDim.x + threadIdx.x;   // B*S*H*64 threads
    if (idx >= (long)BB * SS * HH * 64) return;
    int j = (int)(idx & 63);
    long t = idx >> 6;
    int h = (int)(t & (HH - 1));
    long t2 = t >> 4;
    int s = (int)(t2 & (SS - 1));
    int b = (int)(t2 >> 11);

    float invf = __expf(-((float)j / 64.0f) * logf(10000.0f));
    float ang = (float)s * invf;
    float sn = __sinf(ang), cs = __cosf(ang);

    long src = ((long)b * SS + s) * DD + h * HDIM;
    float q1 = bf2f(q[src + j]), q2 = bf2f(q[src + j + 64]);
    float k1 = bf2f(k[src + j]), k2 = bf2f(k[src + j + 64]);

    long dst = (((long)(b * HH + h)) * SS + s) * HDIM;
    Qr[dst + j]      = f2bf(q1 * cs - q2 * sn);
    Qr[dst + j + 64] = f2bf(q2 * cs + q1 * sn);
    Kr[dst + j]      = f2bf(k1 * cs - k2 * sn);
    Kr[dst + j + 64] = f2bf(k2 * cs + k1 * sn);

    float v1 = bf2f(v[src + j]), v2 = bf2f(v[src + j + 64]);
    long vb = (long)(b * HH + h) * HDIM;
    Vt[(vb + j) * SS + s]      = f2bf(v1);
    Vt[(vb + j + 64) * SS + s] = f2bf(v2);
}

// ---------------- flash attention (WMMA, online softmax + sink) ----------------
// grid: (S/64, B*H), 128 threads (4 waves); wave owns 16 q-rows; k-tiles of 64.
__global__ __launch_bounds__(128) void flash_attn(const __bf16* __restrict__ Q,
                                                  const __bf16* __restrict__ Kt,
                                                  const __bf16* __restrict__ Vt,
                                                  __bf16* __restrict__ O) {
    __shared__ __align__(16) __bf16 Ks[64][HDIM + 8];   // key-major, d contiguous
    __shared__ __align__(16) __bf16 Vs[HDIM][64 + 8];   // d-major, key contiguous
    __shared__ __align__(16) __bf16 Ps[4][16][64 + 8];  // per-wave P tile

    const int bh = blockIdx.y;           // b*H + h
    const int h  = bh & (HH - 1);
    const int b  = bh >> 4;
    const int qt = blockIdx.x;           // q0 = qt*64
    const int tid = threadIdx.x, wave = tid >> 5, lane = tid & 31;
    const int g = lane >> 4, nn = lane & 15;

    // Q fragments for this wave's 16 rows, 4 K-chunks of 32
    const __bf16* Qrow = Q + ((long)bh * SS + qt * 64 + wave * 16 + nn) * HDIM;
    FragU qf[4];
#pragma unroll
    for (int c = 0; c < 4; ++c) {
        qf[c].q[0] = *(const uint4*)(Qrow + c * 32 + 8 * g);
        qf[c].q[1] = *(const uint4*)(Qrow + c * 32 + 16 + 8 * g);
    }

    float mrun[8], lrun[8];
    int qi[8];
#pragma unroll
    for (int v = 0; v < 8; ++v) {
        mrun[v] = 0.0f;           // sink: phantom key with score 0
        lrun[v] = 1.0f;           // exp(0 - 0)
        qi[v] = qt * 64 + wave * 16 + v + 8 * g;
    }
    v8f acc[8] = {};

    const int r  = tid >> 1, c0 = (tid & 1) * 64;      // K-tile staging coords
    const __bf16* Kbase = Kt + ((long)bh * SS + r) * HDIM + c0;
    const __bf16* Vbase = Vt + ((long)bh * HDIM + tid) * SS;

    for (int kt = 0; kt <= qt; ++kt) {
        // stage K/V tiles straight into LDS (async on CDNA5)
        {
            const __bf16* gk = Kbase + (long)kt * 64 * HDIM;
#pragma unroll
            for (int i = 0; i < 8; ++i)
                cp_async_b128(&Ks[r][c0 + i * 8], gk + i * 8);
            const __bf16* gv = Vbase + kt * 64;
#pragma unroll
            for (int i = 0; i < 8; ++i)
                cp_async_b128(&Vs[tid][i * 8], gv + i * 8);
        }
        // prefetch next tile while this one is consumed
        if (kt < qt) {
            __builtin_prefetch(Kbase + (long)(kt + 1) * 64 * HDIM, 0, 0);
            __builtin_prefetch(Vbase + (kt + 1) * 64, 0, 0);
        }
        cp_async_wait();
        __syncthreads();

        // scores: 16 q-rows x 64 keys
        v8f sc[4] = {};
#pragma unroll
        for (int nt = 0; nt < 4; ++nt) {
#pragma unroll
            for (int c = 0; c < 4; ++c) {
                FragU kb;
                const __bf16* rr = &Ks[nt * 16 + nn][c * 32];
                kb.q[0] = *(const uint4*)(rr + 8 * g);
                kb.q[1] = *(const uint4*)(rr + 16 + 8 * g);
                sc[nt] = wmma_bf16(qf[c].f, kb.f, sc[nt]);
            }
        }

        const bool boundary = (kt == qt);
#pragma unroll
        for (int nt = 0; nt < 4; ++nt) {
            int kj = kt * 64 + nt * 16 + nn;
#pragma unroll
            for (int v = 0; v < 8; ++v) {
                float s = sc[nt][v] * SCALEV;
                s = CAPV * fast_tanh(s * (1.0f / CAPV));
                if (boundary && kj > qi[v]) s = -3.0e38f;
                sc[nt][v] = s;
            }
        }

        // per-row max (across 4 n-tiles, then across the 16-lane half)
        float fac[8];
#pragma unroll
        for (int v = 0; v < 8; ++v) {
            float m_ = fmaxf(fmaxf(sc[0][v], sc[1][v]), fmaxf(sc[2][v], sc[3][v]));
#pragma unroll
            for (int off = 1; off < 16; off <<= 1)
                m_ = fmaxf(m_, __shfl_xor(m_, off, 32));
            float mnew = fmaxf(mrun[v], m_);
            fac[v] = __expf(mrun[v] - mnew);
            mrun[v] = mnew;
        }

        // P = exp(s - m), row sums, stash P (bf16) for the PV matmul
        float rs[8] = {0.f, 0.f, 0.f, 0.f, 0.f, 0.f, 0.f, 0.f};
#pragma unroll
        for (int nt = 0; nt < 4; ++nt)
#pragma unroll
            for (int v = 0; v < 8; ++v) {
                float p = __expf(sc[nt][v] - mrun[v]);
                Ps[wave][v + 8 * g][nt * 16 + nn] = f2bf(p);
                rs[v] += p;
            }
#pragma unroll
        for (int v = 0; v < 8; ++v) {
#pragma unroll
            for (int off = 1; off < 16; off <<= 1)
                rs[v] += __shfl_xor(rs[v], off, 32);
            lrun[v] = lrun[v] * fac[v] + rs[v];
        }

        // rescale accumulators
#pragma unroll
        for (int dt = 0; dt < 8; ++dt)
#pragma unroll
            for (int v = 0; v < 8; ++v)
                acc[dt][v] *= fac[v];

        // PV: out(16x128) += P(16x64) x V(64x128)
#pragma unroll
        for (int dt = 0; dt < 8; ++dt) {
#pragma unroll
            for (int kc = 0; kc < 2; ++kc) {
                FragU pa, vb;
                const __bf16* pr = &Ps[wave][nn][kc * 32];
                pa.q[0] = *(const uint4*)(pr + 8 * g);
                pa.q[1] = *(const uint4*)(pr + 16 + 8 * g);
                const __bf16* vr = &Vs[dt * 16 + nn][kc * 32];
                vb.q[0] = *(const uint4*)(vr + 8 * g);
                vb.q[1] = *(const uint4*)(vr + 16 + 8 * g);
                acc[dt] = wmma_bf16(pa.f, vb.f, acc[dt]);
            }
        }
        __syncthreads();
    }

    // epilogue: O[b*S + s][h*128 + d] = acc / l  (bf16 for the Wo GEMM)
#pragma unroll
    for (int dt = 0; dt < 8; ++dt)
#pragma unroll
        for (int v = 0; v < 8; ++v) {
            long row = (long)b * SS + qt * 64 + wave * 16 + v + 8 * g;
            O[row * DD + h * HDIM + dt * 16 + nn] = f2bf(acc[dt][v] / lrun[v]);
        }
}

// ---------------- host orchestration ----------------
extern "C" void kernel_launch(void* const* d_in, const int* in_sizes, int n_in,
                              void* d_out, int out_size, void* d_ws, size_t ws_size,
                              hipStream_t stream) {
    const float* x  = (const float*)d_in[0];
    const float* Wq = (const float*)d_in[1];
    const float* Wk = (const float*)d_in[2];
    const float* Wv = (const float*)d_in[3];
    const float* Wo = (const float*)d_in[4];

    char* ws = (char*)d_ws;
    const long nX = (long)BB * SS * DD;      // 8,388,608
    const long nW = (long)DD * DD;           // 4,194,304
    long off = 0;
    __bf16* xbf  = (__bf16*)(ws + off); off += nX * 2;
    __bf16* WqT  = (__bf16*)(ws + off); off += nW * 2;
    __bf16* WkT  = (__bf16*)(ws + off); off += nW * 2;
    __bf16* WvT  = (__bf16*)(ws + off); off += nW * 2;
    __bf16* WoT  = (__bf16*)(ws + off); off += nW * 2;
    __bf16* qp   = (__bf16*)(ws + off); off += nX * 2;
    __bf16* kp   = (__bf16*)(ws + off); off += nX * 2;
    __bf16* vp   = (__bf16*)(ws + off); off += nX * 2;
    __bf16* Qr   = (__bf16*)(ws + off); off += nX * 2;
    __bf16* Kr   = (__bf16*)(ws + off); off += nX * 2;
    __bf16* Vt   = (__bf16*)(ws + off); off += nX * 2;
    __bf16* attn = (__bf16*)(ws + off); off += nX * 2;
    (void)ws_size; (void)in_sizes; (void)n_in; (void)out_size;

    // 1) convert activations / transpose-convert weights to bf16
    cvt_f32_bf16<<<(int)((nX + 255) / 256), 256, 0, stream>>>(x, xbf, nX);
    transpose_cvt<<<(int)((nW + 255) / 256), 256, 0, stream>>>(Wq, WqT);
    transpose_cvt<<<(int)((nW + 255) / 256), 256, 0, stream>>>(Wk, WkT);
    transpose_cvt<<<(int)((nW + 255) / 256), 256, 0, stream>>>(Wv, WvT);
    transpose_cvt<<<(int)((nW + 255) / 256), 256, 0, stream>>>(Wo, WoT);

    // 2) Q/K/V projections (bf16 WMMA GEMMs), block tile 128x256
    dim3 ggrid(DD / 256, (BB * SS) / 128);   // (8, 32)
    gemm_bf16_nt<false><<<ggrid, 256, 0, stream>>>(xbf, WqT, qp, BB * SS, DD, DD);
    gemm_bf16_nt<false><<<ggrid, 256, 0, stream>>>(xbf, WkT, kp, BB * SS, DD, DD);
    gemm_bf16_nt<false><<<ggrid, 256, 0, stream>>>(xbf, WvT, vp, BB * SS, DD, DD);

    // 3) RoPE + repack (Q,K rotated; V transposed per head)
    long nR = (long)BB * SS * HH * 64;
    rope_pack<<<(int)((nR + 255) / 256), 256, 0, stream>>>(qp, kp, vp, Qr, Kr, Vt);

    // 4) flash attention
    dim3 agrid(SS / 64, BB * HH);            // (32, 32)
    flash_attn<<<agrid, 128, 0, stream>>>(Qr, Kr, Vt, attn);

    // 5) output projection -> fp32 d_out
    gemm_bf16_nt<true><<<ggrid, 256, 0, stream>>>(attn, WoT, d_out, BB * SS, DD, DD);
}
